// Caps_36215164240532
// MI455X (gfx1250) — matrix operand order repaired and checked by reading, not compiled, not run on hardware
//
#include <hip/hip_runtime.h>
#include <hip/hip_bf16.h>

typedef __bf16 bf16;
typedef __attribute__((ext_vector_type(16))) __bf16 v16bf;
typedef __attribute__((ext_vector_type(4)))  __bf16 v4bf;
typedef __attribute__((ext_vector_type(8)))  float  v8f;
typedef __attribute__((ext_vector_type(4)))  float  v4f;
typedef __attribute__((ext_vector_type(4)))  int    v4i;

#define D    96
#define T    288
#define H1   120
#define SPB  64    // samples per block
#define ROWS 128   // token rows per block (2 per sample)
#define QS   132   // sQKV column stride (pad 128 -> 132: 66-DWORD stride, conflict-free)

// workspace layout (bytes):
//   f32 affines: scale1[288] shift1[288] scale2[96] shift2[96] scale3[96] shift3[96]
//   bf16 Wqkv fragments [3*18*32*16] at byte 3840   (WMMA B-frag order)
//   bf16 Wc   fragments [3*6*32*16]  at byte 59136  (Wc = W1@W2 collapsed)
#define WS_WQB_OFF 3840
#define WS_WCB_OFF 59136

#if __has_builtin(__builtin_amdgcn_global_load_async_to_lds_b128)
#define ASYNC_STAGE 1
#define GPTR(p) ((__attribute__((address_space(1))) v4i*)(v4i*)(p))
#define LPTR(p) ((__attribute__((address_space(3))) v4i*)(v4i*)(p))
#endif

__device__ __forceinline__ void wait_async_all() {
#if defined(ASYNC_STAGE)
#if __has_builtin(__builtin_amdgcn_s_wait_asynccnt)
    __builtin_amdgcn_s_wait_asynccnt(0);
#else
    asm volatile("s_wait_asynccnt 0x0" ::: "memory");
#endif
#endif
}

// bf16 (packed in a dword) -> f32 without cvt ops
__device__ __forceinline__ float bflo(unsigned v) { return __uint_as_float(v << 16); }
__device__ __forceinline__ float bfhi(unsigned v) { return __uint_as_float(v & 0xffff0000u); }

__device__ __forceinline__ void packA(v16bf& a, const float4 f0, const float4 f1,
                                      const float4 f2, const float4 f3) {
    a[0]  = (bf16)f0.x; a[1]  = (bf16)f0.y; a[2]  = (bf16)f0.z; a[3]  = (bf16)f0.w;
    a[4]  = (bf16)f1.x; a[5]  = (bf16)f1.y; a[6]  = (bf16)f1.z; a[7]  = (bf16)f1.w;
    a[8]  = (bf16)f2.x; a[9]  = (bf16)f2.y; a[10] = (bf16)f2.z; a[11] = (bf16)f2.w;
    a[12] = (bf16)f3.x; a[13] = (bf16)f3.y; a[14] = (bf16)f3.z; a[15] = (bf16)f3.w;
}

// ---------------------------------------------------------------------------
// Prep kernel: fold biases + batchnorm into per-column affines, convert
// weights to bf16 in *WMMA B-fragment order*, collapse linear MLP W1@W2 -> Wc.
// ---------------------------------------------------------------------------
__global__ void prep_kernel(const float* __restrict__ Wqkv, const float* __restrict__ bqkv,
                            const float* __restrict__ g1, const float* __restrict__ b1n,
                            const float* __restrict__ m1, const float* __restrict__ v1,
                            const float* __restrict__ W1, const float* __restrict__ b1,
                            const float* __restrict__ W2, const float* __restrict__ b2,
                            const float* __restrict__ g2, const float* __restrict__ b2n,
                            const float* __restrict__ m2, const float* __restrict__ v2,
                            const float* __restrict__ g3, const float* __restrict__ b3n,
                            const float* __restrict__ m3, const float* __restrict__ v3,
                            float* __restrict__ wsF, bf16* __restrict__ wqb,
                            bf16* __restrict__ wcb)
{
    const int tid = threadIdx.x;
    for (int c = tid; c < T; c += 256) {
        float sc = g1[c] * rsqrtf(v1[c] + 1e-3f);
        wsF[c]       = sc;
        wsF[288 + c] = (bqkv[c] - m1[c]) * sc + b1n[c];
    }
    for (int c = tid; c < D; c += 256) {
        float sc = g2[c] * rsqrtf(v2[c] + 1e-3f);
        wsF[576 + c] = sc;
        wsF[672 + c] = b2n[c] - m2[c] * sc;
    }
    for (int c = tid; c < D; c += 256) {
        float bc = b2[c];
        for (int k = 0; k < H1; ++k) bc += b1[k] * W2[k * D + c];
        float sc = g3[c] * rsqrtf(v3[c] + 1e-3f);
        wsF[768 + c] = sc;
        wsF[864 + c] = b3n[c] - m3[c] * sc + bc * sc;
    }
    for (int f = tid; f < 3 * 18 * 32 * 16; f += 256) {
        const int i    = f & 15;
        const int lane = (f >> 4) & 31;
        const int rest = f >> 9;
        const int nt   = rest % 18, ks = rest / 18;
        const int k = ks * 32 + ((lane >> 4) << 3) + (i & 7) + ((i >> 3) << 4);
        const int n = nt * 16 + (lane & 15);
        wqb[f] = (bf16)Wqkv[k * T + n];
    }
    for (int f = tid; f < 3 * 6 * 32 * 16; f += 256) {
        const int i    = f & 15;
        const int lane = (f >> 4) & 31;
        const int rest = f >> 9;
        const int nt   = rest % 6, ks = rest / 6;
        const int k = ks * 32 + ((lane >> 4) << 3) + (i & 7) + ((i >> 3) << 4);
        const int n = nt * 16 + (lane & 15);
        float acc = 0.f;
        for (int j = 0; j < H1; ++j) acc += W1[k * H1 + j] * W2[j * D + n];
        wcb[f] = (bf16)acc;
    }
}

// ---------------------------------------------------------------------------
// Fully fused main kernel: 64 samples (128 token rows) per 256-thread block.
// ---------------------------------------------------------------------------
__global__ __launch_bounds__(256, 1) void fused_kernel(
    const float* __restrict__ inputs,
    const float* __restrict__ wsF,
    const bf16*  __restrict__ wqb,
    const bf16*  __restrict__ wcb,
    const float* __restrict__ Wgi, const float* __restrict__ bgi,
    const float* __restrict__ Wgm, const float* __restrict__ bgm,
    float* __restrict__ out)
{
    __shared__ __align__(32) float sX[ROWS * D];     // 49152 B  input tile (f32)
    __shared__ __align__(32) bf16  sWq[D * T];       // 55296 B  Wqkv frags
    __shared__ __align__(32) bf16  sQKV[T * QS];     // 76032 B  qkv bf16, column-major
    __shared__ __align__(32) bf16  sWc[D * D];       // 18432 B  Wc frags
    __shared__ __align__(32) float sNM[SPB * D];     // 24576 B  BN2(x+attn), token 0
    __shared__ __align__(32) float sNM3[SPB * D];    // 24576 B  BN3(mlp+nm), token 0
    __shared__ float sGates[SPB * 2];                //   512 B

    const int tid  = threadIdx.x;
    const int lane = tid & 31;
    const int wave = tid >> 5;
    const int l15  = lane & 15;
    const int hi8  = (lane >> 4) << 3;   // 0 or 8  (A frag K-half / D-row half)

    // -------- Phase 0: stage input tile + weights into LDS (async engine) --------
    {
        const float4* xs = (const float4*)(inputs + (size_t)blockIdx.x * (ROWS * D));
        float4* xd = (float4*)sX;
        const uint4* s1 = (const uint4*)wqb; uint4* d1 = (uint4*)sWq;
        const uint4* s2 = (const uint4*)wcb; uint4* d2 = (uint4*)sWc;
#if defined(ASYNC_STAGE)
        for (int i = tid; i < ROWS * D / 4; i += 256)
            __builtin_amdgcn_global_load_async_to_lds_b128(
                GPTR(xs + i), LPTR(xd + i), 0, 0);
        for (int i = tid; i < D * T * 2 / 16; i += 256)
            __builtin_amdgcn_global_load_async_to_lds_b128(
                GPTR(s1 + i), LPTR(d1 + i), 0, 0);
        for (int i = tid; i < D * D * 2 / 16; i += 256)
            __builtin_amdgcn_global_load_async_to_lds_b128(
                GPTR(s2 + i), LPTR(d2 + i), 0, 0);
        wait_async_all();
#else
        for (int i = tid; i < ROWS * D / 4; i += 256) xd[i] = xs[i];
        for (int i = tid; i < D * T * 2 / 16; i += 256) d1[i] = s1[i];
        for (int i = tid; i < D * D * 2 / 16; i += 256) d2[i] = s2[i];
#endif
    }
    __syncthreads();

    // -------- Phase 1: qkv = BN1(X @ Wqkv + bqkv)  [WMMA bf16] --------
    {
        v8f acc[18] = {};
        const int mbase = wave * 16;
        #pragma unroll
        for (int ks = 0; ks < 3; ++ks) {
            const int kS = ks * 32;
            const float4* ap = (const float4*)(&sX[(mbase + l15) * D + kS + hi8]);
            v16bf a;
            packA(a, ap[0], ap[1], ap[4], ap[5]);   // K 0..7 and 16..23 (+half)
            #pragma unroll
            for (int nt = 0; nt < 18; ++nt) {
                const v16bf b = *(const v16bf*)(&sWq[((ks * 18 + nt) * 32 + lane) * 16]);
                acc[nt] = __builtin_amdgcn_wmma_f32_16x16x32_bf16(
                    false, a, false, b, (short)0, acc[nt], false, false);
            }
        }
        // column-major store: lane's 8 rows are contiguous -> 2x ds_store_b64
        #pragma unroll
        for (int nt = 0; nt < 18; ++nt) {
            const int col = nt * 16 + l15;
            const float sc = wsF[col];
            const float sh = wsF[288 + col];
            v4bf lo, hi;
            #pragma unroll
            for (int r = 0; r < 4; ++r) lo[r] = (bf16)(acc[nt][r] * sc + sh);
            #pragma unroll
            for (int r = 0; r < 4; ++r) hi[r] = (bf16)(acc[nt][r + 4] * sc + sh);
            bf16* dst = &sQKV[col * QS + mbase + hi8];
            *(v4bf*)dst       = lo;
            *(v4bf*)(dst + 4) = hi;
        }
    }
    __syncthreads();

    // -------- Phase 2: attention (seq=2, only q-token 0 needed) + BN2 --------
    {
        for (int j = 0; j < 24; ++j) {
            const int u = wave * 24 + j;          // 192 units = 64 samples * 3 heads
            const int s = u / 3, h = u - 3 * s;
            const int r0 = 2 * s;                 // token-0 row; token-1 adjacent
            const int qc = h * 96;
            const unsigned qr = *(const unsigned short*)&sQKV[(qc + lane) * QS + r0];
            const unsigned kk = *(const unsigned*)&sQKV[(qc + 32 + lane) * QS + r0];
            const unsigned vv = *(const unsigned*)&sQKV[(qc + 64 + lane) * QS + r0];
            const float q0 = __uint_as_float(qr << 16);
            const float k0 = bflo(kk), k1 = bfhi(kk);
            const float v0 = bflo(vv), v1 = bfhi(vv);
            float p0 = q0 * k0, p1 = q0 * k1;
            #pragma unroll
            for (int off = 16; off > 0; off >>= 1) {
                p0 += __shfl_xor(p0, off, 32);
                p1 += __shfl_xor(p1, off, 32);
            }
            const float mx  = fmaxf(p0, p1);
            const float e0  = __expf(p0 - mx), e1 = __expf(p1 - mx);
            const float att = (e0 * v0 + e1 * v1) / (e0 + e1);
            const int   c   = h * 32 + lane;      // attended column after head transpose
            sNM[s * D + c] = (sX[r0 * D + c] + att) * wsF[576 + c] + wsF[672 + c];
        }
    }
    __syncthreads();

    // -------- Phase 3: nm3 = BN3(nm @ Wc + bc + nm)  [WMMA bf16] --------
    {
        v8f acc2[3] = {};
        const int mt    = wave >> 1;
        const int nbase = (wave & 1) * 3;
        #pragma unroll
        for (int ks = 0; ks < 3; ++ks) {
            const int kS = ks * 32;
            const float4* ap = (const float4*)(&sNM[(mt * 16 + l15) * D + kS + hi8]);
            v16bf a;
            packA(a, ap[0], ap[1], ap[4], ap[5]);
            #pragma unroll
            for (int t = 0; t < 3; ++t) {
                const v16bf b =
                    *(const v16bf*)(&sWc[((ks * 6 + nbase + t) * 32 + lane) * 16]);
                acc2[t] = __builtin_amdgcn_wmma_f32_16x16x32_bf16(
                    false, a, false, b, (short)0, acc2[t], false, false);
            }
        }
        #pragma unroll
        for (int t = 0; t < 3; ++t) {
            const int col = (nbase + t) * 16 + l15;
            const float sc = wsF[768 + col], sh = wsF[864 + col];
            #pragma unroll
            for (int r = 0; r < 8; ++r) {
                const int srow = mt * 16 + r + hi8;
                sNM3[srow * D + col] = (acc2[t][r] + sNM[srow * D + col]) * sc + sh;
            }
        }
    }
    __syncthreads();

    // -------- Phase 4: LSTM-style gates (two 96->2 matvecs per sample) --------
    {
        const int sLoc = wave * 8 + (lane >> 2);  // 8 samples per wave
        const int part = lane & 3;                // 4 lanes cooperate per sample
        float pi0 = 0.f, pi1 = 0.f, pm0 = 0.f, pm1 = 0.f;
        const float* xin = &sX[(2 * sLoc + 1) * D];  // token 1 = x_in
        const float* mem = &sX[(2 * sLoc) * D];      // token 0 = memory
        for (int d = part; d < D; d += 4) {
            const float xv = xin[d];
            const float mt = tanhf(mem[d]);
            pi0 += xv * Wgi[d * 2];  pi1 += xv * Wgi[d * 2 + 1];
            pm0 += mt * Wgm[d * 2];  pm1 += mt * Wgm[d * 2 + 1];
        }
        pi0 += __shfl_xor(pi0, 1, 32); pi0 += __shfl_xor(pi0, 2, 32);
        pi1 += __shfl_xor(pi1, 1, 32); pi1 += __shfl_xor(pi1, 2, 32);
        pm0 += __shfl_xor(pm0, 1, 32); pm0 += __shfl_xor(pm0, 2, 32);
        pm1 += __shfl_xor(pm1, 1, 32); pm1 += __shfl_xor(pm1, 2, 32);
        if (part == 0) {
            const float gi = pi0 + pm0 + bgi[0] + bgm[0];          // INPUT_BIAS = 0
            const float gf = pi1 + pm1 + bgi[1] + bgm[1] + 1.0f;   // FORGET_BIAS = 1
            sGates[2 * sLoc]     = 1.f / (1.f + __expf(-gi));
            sGates[2 * sLoc + 1] = 1.f / (1.f + __expf(-gf));
        }
    }
    __syncthreads();

    // -------- Phase 5: next_memory = ig*tanh(nm1) + fg*memory --------
    // v4f + non-temporal stores: output is a 100 MB write-once stream.
    {
        float* oBase = out + (size_t)blockIdx.x * (ROWS * D);
        const v4f* nm34 = (const v4f*)sNM3;
        for (int idx = tid; idx < SPB * D / 4; idx += 256) {
            const int s = idx / (D / 4), q = idx - s * (D / 4);
            const float ig = sGates[2 * s], fg = sGates[2 * s + 1];
            const v4f nv  = nm34[idx];
            const v4f mem = *(const v4f*)&sX[2 * s * D + 4 * q];
            v4f nx;
            nx.x = ig * tanhf(nv.x) + fg * mem.x;
            nx.y = ig * tanhf(nv.y) + fg * mem.y;
            nx.z = ig * tanhf(nv.z) + fg * mem.z;
            nx.w = ig * tanhf(nv.w) + fg * mem.w;
            __builtin_nontemporal_store(nx, (v4f*)(oBase + s * 2 * D + 4 * q));
            __builtin_nontemporal_store(nx, (v4f*)(oBase + s * 2 * D + D + 4 * q));
        }
    }
}

// ---------------------------------------------------------------------------
extern "C" void kernel_launch(void* const* d_in, const int* in_sizes, int n_in,
                              void* d_out, int out_size, void* d_ws, size_t ws_size,
                              hipStream_t stream)
{
    const float* inputs = (const float*)d_in[0];
    const float* Wqkv   = (const float*)d_in[1];
    const float* bqkv   = (const float*)d_in[2];
    const float* bn1_g  = (const float*)d_in[3];
    const float* bn1_b  = (const float*)d_in[4];
    const float* bn1_m  = (const float*)d_in[5];
    const float* bn1_v  = (const float*)d_in[6];
    const float* W1     = (const float*)d_in[7];
    const float* b1     = (const float*)d_in[8];
    const float* W2     = (const float*)d_in[9];
    const float* b2     = (const float*)d_in[10];
    const float* bn2_g  = (const float*)d_in[11];
    const float* bn2_b  = (const float*)d_in[12];
    const float* bn2_m  = (const float*)d_in[13];
    const float* bn2_v  = (const float*)d_in[14];
    const float* bn3_g  = (const float*)d_in[15];
    const float* bn3_b  = (const float*)d_in[16];
    const float* bn3_m  = (const float*)d_in[17];
    const float* bn3_v  = (const float*)d_in[18];
    const float* Wgi    = (const float*)d_in[19];
    const float* bgi    = (const float*)d_in[20];
    const float* Wgm    = (const float*)d_in[21];
    const float* bgm    = (const float*)d_in[22];

    float* wsF = (float*)d_ws;
    bf16*  wqb = (bf16*)((char*)d_ws + WS_WQB_OFF);
    bf16*  wcb = (bf16*)((char*)d_ws + WS_WCB_OFF);

    prep_kernel<<<1, 256, 0, stream>>>(Wqkv, bqkv, bn1_g, bn1_b, bn1_m, bn1_v,
                                       W1, b1, W2, b2, bn2_g, bn2_b, bn2_m, bn2_v,
                                       bn3_g, bn3_b, bn3_m, bn3_v, wsF, wqb, wcb);

    const int B = in_sizes[0] / (2 * D);     // 131072
    const int blocks = B / SPB;              // 2048
    fused_kernel<<<blocks, 256, 0, stream>>>(inputs, wsF, wqb, wcb,
                                             Wgi, bgi, Wgm, bgm, (float*)d_out);
}
